// GATLayer_84018150245122
// MI455X (gfx1250) — matrix operand order repaired and checked by reading, not compiled, not run on hardware
//
#include <hip/hip_runtime.h>

typedef __bf16 bf16_t;
typedef __attribute__((ext_vector_type(16))) __bf16 v16bf;
typedef __attribute__((ext_vector_type(8)))  __bf16 v8bf;
typedef __attribute__((ext_vector_type(8)))  float   v8f;
typedef __attribute__((ext_vector_type(4)))  float   v4f;
typedef __attribute__((ext_vector_type(2)))  int     v2i;

#define NND   4096
#define FIN   128
#define FOUT  64
#define NH    8
#define NEG_SLOPE 0.2f

// ---------------------------------------------------- CDNA5 async global->LDS
__device__ __forceinline__ void async_copy_b64(const float* gsrc, float* ldst) {
#if __has_builtin(__builtin_amdgcn_global_load_async_to_lds_b64)
    float* gs = const_cast<float*>(gsrc);
    __attribute__((address_space(1))) v2i* gv =
        (__attribute__((address_space(1))) v2i*)gs;
    __attribute__((address_space(3))) v2i* lv =
        (__attribute__((address_space(3))) v2i*)ldst;
    __builtin_amdgcn_global_load_async_to_lds_b64(gv, lv, 0, 0);
#else
    unsigned loff = (unsigned)(size_t)(__attribute__((address_space(3))) void*)ldst;
    asm volatile("global_load_async_to_lds_b64 %0, %1, off"
                 :: "v"(loff), "v"(gsrc) : "memory");
#endif
}

__device__ __forceinline__ void wait_async0() {
#if __has_builtin(__builtin_amdgcn_s_wait_asynccnt)
    __builtin_amdgcn_s_wait_asynccnt(0);
#else
    asm volatile("s_wait_asynccnt 0x0" ::: "memory");
#endif
}

// ---------------------------------------------------------------- pack kernels
__global__ void pack_x(const float* __restrict__ x, bf16_t* __restrict__ xb) {
    int i = blockIdx.x * 256 + threadIdx.x;
    xb[i] = (bf16_t)x[i];
}

// W[k][c] (128 x 1024 bf16): c<512 -> proj[h][k][o] (h=c/64,o=c%64); c>=512 -> skip_w[c-512][k]
__global__ void pack_w(const float* __restrict__ proj, const float* __restrict__ skip_w,
                       bf16_t* __restrict__ wb) {
    int i = blockIdx.x * 256 + threadIdx.x;   // i = k*1024 + c
    int k = i >> 10, c = i & 1023;
    float v;
    if (c < 512) { int h = c >> 6, o = c & 63; v = proj[(h * FIN + k) * FOUT + o]; }
    else         { v = skip_w[(size_t)(c - 512) * FIN + k]; }
    wb[i] = (bf16_t)v;
}

// ------------------------------------------------- projection GEMM (WMMA bf16)
// Y[4096,1024] = x_bf16[4096,128] @ W_bf16[128,1024], fp32 accumulate.
__global__ __launch_bounds__(256) void proj_gemm(const bf16_t* __restrict__ xb,
                                                 const bf16_t* __restrict__ wb,
                                                 float* __restrict__ Y) {
    int wave = threadIdx.x >> 5;
    int lane = threadIdx.x & 31;
    int mb = blockIdx.x & 255;      // 256 M blocks of 16 rows
    int ng = blockIdx.x >> 8;       // 8 N groups x 8 waves
    int n0 = (ng * 8 + wave) * 16;
    int i0 = mb * 16;
    int row = lane & 15;
    int hi  = lane >> 4;

    v8f acc = {};
#pragma unroll
    for (int kb = 0; kb < 4; ++kb) {
        // A: 16-bit 16x32 layout — lane row M=lane&15; lo half K{0..7,16..23}, hi half +8
        int kA = kb * 32 + hi * 8;
        const bf16_t* ap = xb + (size_t)(i0 + row) * FIN + kA;
        v8bf a_lo = *(const v8bf*)(ap);
        v8bf a_hi = *(const v8bf*)(ap + 16);
        v16bf A = __builtin_shufflevector(a_lo, a_hi, 0,1,2,3,4,5,6,7,8,9,10,11,12,13,14,15);
        // B: lane col N=n0+(lane&15); K in table order (same as A)
        v16bf B;
#pragma unroll
        for (int s = 0; s < 16; ++s)
            B[s] = wb[(size_t)(kb * 32 + hi * 8 + (s & 7) + (s >> 3) * 16) * 1024 + n0 + row];
        acc = __builtin_amdgcn_wmma_f32_16x16x32_bf16(false, A, false, B, (short)0, acc,
                                                      false, false);
    }
    // C/D layout: lane col N=lane&15; VGPR r -> row r (+8 for hi half)
    float* yp = Y + (size_t)(i0 + hi * 8) * 1024 + n0 + row;
#pragma unroll
    for (int r = 0; r < 8; ++r) yp[(size_t)r * 1024] = acc[r];
}

// ------------------------- per-head bf16 copy of h + attention score pre-dots
__global__ void head_prep(const float* __restrict__ Y, const float* __restrict__ ssrc_w,
                          const float* __restrict__ sdst_w, bf16_t* __restrict__ hb,
                          float* __restrict__ s_src, float* __restrict__ s_dst) {
    int t = blockIdx.x * blockDim.x + threadIdx.x;   // 0..32767
    int h = t >> 12;
    int n = t & 4095;
    const float* yp = Y + (size_t)n * 1024 + h * 64;
    bf16_t* hp = hb + ((size_t)h * NND + n) * 64;
    float a = 0.f, b = 0.f;
#pragma unroll 8
    for (int o = 0; o < 64; ++o) {
        float v = yp[o];
        hp[o] = (bf16_t)v;
        a += v * ssrc_w[h * 64 + o];
        b += v * sdst_w[h * 64 + o];
    }
    s_src[t] = a;
    s_dst[t] = b;
}

// ------------------------------------- fused flash-softmax attention + output
// Block = 8 waves (one head each) over one 16-row i-block. The 16x32 topology
// tile is DMA'd ONCE per workgroup into LDS via async global->LDS (ASYNCcnt),
// double-buffered so the next tile's copy overlaps this tile's exp+WMMA work.
__global__ __launch_bounds__(256) void gat_attn(const float* __restrict__ topo,
                                                const bf16_t* __restrict__ hb,
                                                const float* __restrict__ s_src,
                                                const float* __restrict__ s_dst,
                                                const float* __restrict__ Y,
                                                float* __restrict__ out) {
    __shared__ float topoS[2][16][36];     // padded rows: spread LDS banks
    __shared__ float red[NH][16][64];

    int head = threadIdx.x >> 5;
    int lane = threadIdx.x & 31;
    int row  = lane & 15;
    int hi   = lane >> 4;
    int base = hi * 8;              // K-offset base of this lane's half (A layout)
    int i0   = blockIdx.x * 16;

    // async tile slice owned by this wave: rows {2w, 2w+1}, one b64 per lane
    int tr = head * 2 + hi;         // tile row this lane copies
    int tc = row * 2;               // tile col (even -> b64 aligned)
    const float* topo_tile0 = topo + (size_t)(i0 + tr) * NND + tc;

    float si = s_src[head * NND + i0 + row];
    float m = -3.0e38f, l = 0.f;
    v8f acc0 = {}, acc1 = {}, acc2 = {}, acc3 = {};

    const float*  sd = s_dst + head * NND;
    const bf16_t* hh = hb + (size_t)head * NND * 64;

    // prologue: stage tile j0=0 into buffer 0
    async_copy_b64(topo_tile0, &topoS[0][tr][tc]);
    wait_async0();
    __syncthreads();

    for (int j0 = 0; j0 < NND; j0 += 32) {
        int buf = (j0 >> 5) & 1;
        // kick off DMA of the next tile into the other buffer
        if (j0 + 32 < NND)
            async_copy_b64(topo_tile0 + (j0 + 32), &topoS[buf ^ 1][tr][tc]);

        // this lane's 16 K offsets: {base..base+7, base+16..base+23} from LDS
        v4f t0 = *(const v4f*)(&topoS[buf][row][base]);
        v4f t1 = *(const v4f*)(&topoS[buf][row][base + 4]);
        v4f t2 = *(const v4f*)(&topoS[buf][row][base + 16]);
        v4f t3 = *(const v4f*)(&topoS[buf][row][base + 20]);
        v4f d0 = *(const v4f*)(sd + j0 + base);
        v4f d1 = *(const v4f*)(sd + j0 + base + 4);
        v4f d2 = *(const v4f*)(sd + j0 + base + 16);
        v4f d3 = *(const v4f*)(sd + j0 + base + 20);

        float tv[16], dv[16];
#pragma unroll
        for (int e = 0; e < 4; ++e) {
            tv[e] = t0[e]; tv[4 + e] = t1[e]; tv[8 + e] = t2[e]; tv[12 + e] = t3[e];
            dv[e] = d0[e]; dv[4 + e] = d1[e]; dv[8 + e] = d2[e]; dv[12 + e] = d3[e];
        }

        float sc[16];
        float bm = -3.0e38f;
#pragma unroll
        for (int e = 0; e < 16; ++e) {
            float s = si + dv[e];
            s = (s >= 0.f) ? s : NEG_SLOPE * s;   // leaky relu on raw score
            s += tv[e];                           // additive -1e9 mask
            sc[e] = s;
            bm = fmaxf(bm, s);
        }
        // row M=lane&15 is split across lanes (L, L+16): pairwise reduce
        bm = fmaxf(bm, __shfl_xor(bm, 16, 32));
        float nm = fmaxf(m, bm);
        float scale = __expf(m - nm);
        m = nm;

        float ps = 0.f;
        v16bf A;
#pragma unroll
        for (int e = 0; e < 16; ++e) {
            float p = __expf(sc[e] - nm);
            ps += p;
            A[e] = (bf16_t)p;                     // element e == A-layout K slot e
        }
        ps += __shfl_xor(ps, 16, 32);
        l = l * scale + ps;

        // rescale accumulators: C VGPR r holds row r+8*hi; fetch that row's scale
#pragma unroll
        for (int r = 0; r < 8; ++r) {
            float s_r = __shfl(scale, r + base, 32);
            acc0[r] *= s_r; acc1[r] *= s_r; acc2[r] *= s_r; acc3[r] *= s_r;
        }

        // B tiles: lane col N=lane&15 per tile; K in table order (same as A)
        v16bf B0, B1, B2, B3;
#pragma unroll
        for (int e = 0; e < 16; ++e) {
            const bf16_t* bp = hh + (size_t)(j0 + base + (e & 7) + (e >> 3) * 16) * 64 + row;
            B0[e] = bp[0]; B1[e] = bp[16]; B2[e] = bp[32]; B3[e] = bp[48];
        }
        acc0 = __builtin_amdgcn_wmma_f32_16x16x32_bf16(false, A, false, B0, (short)0, acc0, false, false);
        acc1 = __builtin_amdgcn_wmma_f32_16x16x32_bf16(false, A, false, B1, (short)0, acc1, false, false);
        acc2 = __builtin_amdgcn_wmma_f32_16x16x32_bf16(false, A, false, B2, (short)0, acc2, false, false);
        acc3 = __builtin_amdgcn_wmma_f32_16x16x32_bf16(false, A, false, B3, (short)0, acc3, false, false);

        // close the double-buffer window: DMA landed + all waves done with buf
        wait_async0();
        __syncthreads();
    }

    // normalize and stage per-head tile in LDS
    float invl = 1.0f / l;
#pragma unroll
    for (int r = 0; r < 8; ++r) {
        float w = __shfl(invl, r + base, 32);
        int rr = r + base;
        red[head][rr][ 0 + row] = acc0[r] * w;
        red[head][rr][16 + row] = acc1[r] * w;
        red[head][rr][32 + row] = acc2[r] * w;
        red[head][rr][48 + row] = acc3[r] * w;
    }
    __syncthreads();

    // cross-head mean + projected skip + final leaky relu
#pragma unroll
    for (int q = 0; q < 4; ++q) {
        int flat = threadIdx.x + q * 256;     // 0..1023 over 16x64 outputs
        int rr = flat >> 6;
        int oo = flat & 63;
        float sum = 0.f;
#pragma unroll
        for (int h2 = 0; h2 < NH; ++h2)
            sum += red[h2][rr][oo] + Y[(size_t)(i0 + rr) * 1024 + 512 + h2 * 64 + oo];
        sum *= (1.0f / NH);
        sum = (sum >= 0.f) ? sum : NEG_SLOPE * sum;
        out[(size_t)(i0 + rr) * 64 + oo] = sum;
    }
}

// ---------------------------------------------------------------------- launch
extern "C" void kernel_launch(void* const* d_in, const int* in_sizes, int n_in,
                              void* d_out, int out_size, void* d_ws, size_t ws_size,
                              hipStream_t stream) {
    const float* x      = (const float*)d_in[0];   // [4096,128]
    const float* topo   = (const float*)d_in[1];   // [4096,4096]
    const float* proj   = (const float*)d_in[2];   // [8,128,64]
    const float* ssrc_w = (const float*)d_in[3];   // [8,64]
    const float* sdst_w = (const float*)d_in[4];   // [8,64]
    const float* skip_w = (const float*)d_in[5];   // [512,128]

    char* ws = (char*)d_ws;
    bf16_t* xb    = (bf16_t*)(ws);               //  1,048,576 B
    bf16_t* wbm   = (bf16_t*)(ws +  1048576);    //    262,144 B
    float*  Y     = (float* )(ws +  1310720);    // 16,777,216 B
    bf16_t* hb    = (bf16_t*)(ws + 18087936);    //  4,194,304 B
    float*  s_src = (float* )(ws + 22282240);    //    131,072 B
    float*  s_dst = (float* )(ws + 22413312);    //    131,072 B
    float*  out   = (float*)d_out;               // [4096,64]

    pack_x<<<(NND * FIN) / 256, 256, 0, stream>>>(x, xb);
    pack_w<<<(FIN * 1024) / 256, 256, 0, stream>>>(proj, skip_w, wbm);
    proj_gemm<<<2048, 256, 0, stream>>>(xb, wbm, Y);
    head_prep<<<(NH * NND) / 256, 256, 0, stream>>>(Y, ssrc_w, sdst_w, hb, s_src, s_dst);
    gat_attn<<<NND / 16, 256, 0, stream>>>(topo, hb, s_src, s_dst, Y, out);
}
